// up_block_batch_26388279067306
// MI455X (gfx1250) — matrix operand order repaired and checked by reading, not compiled, not run on hardware
//
#include <hip/hip_runtime.h>
#include <hip/hip_bf16.h>

typedef __attribute__((ext_vector_type(16))) __bf16 v16bf;
typedef __attribute__((ext_vector_type(8)))  __bf16 v8bf;
typedef __attribute__((ext_vector_type(8)))  float  v8f;

union ABFrag { v16bf v; v8bf h[2]; };

__device__ __forceinline__ v8f wmma_bf16(const ABFrag& a, const ABFrag& b, v8f c) {
  return __builtin_amdgcn_wmma_f32_16x16x32_bf16(false, a.v, false, b.v,
                                                 (short)0, c, false, false);
}

// ---------------------------------------------------------------------------
// Transpose a K x N f32 weight matrix into N x K bf16 (row per output channel)
// ---------------------------------------------------------------------------
__global__ void transpose_w_kernel(const float* __restrict__ src,
                                   __bf16* __restrict__ dst, int K, int N) {
  int t = blockIdx.x * blockDim.x + threadIdx.x;
  if (t >= K * N) return;
  int k = t / N;
  int n = t - k * N;
  dst[(size_t)n * K + k] = (__bf16)src[t];
}

// ---------------------------------------------------------------------------
// h = x1 @ up_W + up_b   (M x 128) x (128 x 448), bf16 WMMA, f32 accumulate.
// One block = 16 rows; 4 waves cycle over the 28 N-tiles of 16.
// Bias is pre-seeded into the accumulator (bias depends only on out column,
// which is constant per lane in the C/D layout).
// ---------------------------------------------------------------------------
__global__ __launch_bounds__(128) void upproj_wmma_kernel(
    const float* __restrict__ x1, const __bf16* __restrict__ wt /*448x128*/,
    const float* __restrict__ upb, float* __restrict__ h, int M) {
  __shared__ __bf16 tileA[16 * 128];
  const int tid = threadIdx.x;
  const int rowBase = blockIdx.x * 16;

  // stage 16x128 f32 -> bf16 into LDS (clamp tail rows)
  for (int t = tid; t < 16 * 128; t += 128) {
    int m = t >> 7, c = t & 127;
    int q = rowBase + m; if (q > M - 1) q = M - 1;
    tileA[t] = (__bf16)x1[(size_t)q * 128 + c];
  }
  __syncthreads();

  const int lane = tid & 31, wave = tid >> 5;
  const int col = lane & 15, hl = lane >> 4;
  const __bf16* arow = tileA + col * 128;
  const bool full = (rowBase + 16 <= M);   // block-uniform fast path

  for (int nt = wave; nt < 28; nt += 4) {
    const float bv = upb[nt * 16 + col];
    v8f acc = {bv, bv, bv, bv, bv, bv, bv, bv};
    const __bf16* brow = wt + (size_t)(nt * 16 + col) * 128 + hl * 16;
    for (int k0 = 0; k0 < 128; k0 += 32) {
      ABFrag a, b;
      a.h[0] = *(const v8bf*)(arow + k0 + hl * 16);
      a.h[1] = *(const v8bf*)(arow + k0 + hl * 16 + 8);
      b.h[0] = *(const v8bf*)(brow + k0);
      b.h[1] = *(const v8bf*)(brow + k0 + 8);
      acc = wmma_bf16(a, b, acc);
    }
    float* out = h + (size_t)rowBase * 448 + nt * 16 + col;
    if (full) {
      #pragma unroll
      for (int r = 0; r < 8; ++r)
        out[(size_t)(r + hl * 8) * 448] = acc[r];
    } else {
      #pragma unroll
      for (int r = 0; r < 8; ++r) {
        int m = r + hl * 8;
        if (rowBase + m < M) out[(size_t)m * 448] = acc[r];
      }
    }
  }
}

// ---------------------------------------------------------------------------
// Build concat activation xcat (B*Nh rows x 128 ch, bf16):
//   ch 0..63  : top gather (n < Nr) or mean-of-2 down gather (n >= Nr) from h
//   ch 64..127: x2
// ---------------------------------------------------------------------------
__global__ void build_xcat_kernel(const float* __restrict__ h,
                                  const float* __restrict__ x2,
                                  const int* __restrict__ topIdx,
                                  const int* __restrict__ downIdx,
                                  __bf16* __restrict__ xcat,
                                  int Nr, int Nh, int total) {
  int t = blockIdx.x * blockDim.x + threadIdx.x;
  if (t >= total) return;
  int c = t & 127;
  int q = t >> 7;                  // q = b*Nh + n
  int b = q / Nh;
  int n = q - b * Nh;
  float v;
  if (c < 64) {
    size_t hbase = (size_t)b * Nr * 7;
    if (n < Nr) {
      int s = topIdx[n];
      v = h[(hbase + s) * 64 + c];
    } else {
      int i = n - Nr;
      int s0 = downIdx[2 * i], s1 = downIdx[2 * i + 1];
      v = 0.5f * (h[(hbase + s0) * 64 + c] + h[(hbase + s1) * 64 + c]);
    }
  } else {
    v = x2[(size_t)q * 64 + (c - 64)];
  }
  xcat[t] = (__bf16)v;
}

// ---------------------------------------------------------------------------
// Fused gather + GEMM mesh conv: y[q, 0:64] = concat_j x[neigh[n,j]] @ W + b
// Block = 32 output rows (two 16-row subtiles) so each B fragment feeds two
// WMMAs; 4 waves cover the 64 output channels. A tile gathered into LDS.
// ---------------------------------------------------------------------------
template <int C>
__global__ __launch_bounds__(128) void meshconv_wmma_kernel(
    const __bf16* __restrict__ x, const int* __restrict__ neigh,
    const __bf16* __restrict__ wt /* 64 x 7C */, const float* __restrict__ bias,
    float* __restrict__ y, int Nh, int R) {
  constexpr int K = 7 * C;
  __shared__ __bf16 tileA[32 * K];
  const int tid = threadIdx.x;
  const int rowBase = blockIdx.x * 32;

  constexpr int CH = C / 8;             // 16-byte chunks per neighbor segment
  constexpr int TOT = 32 * 7 * CH;      // multiple of 128 for C in {64,128}
  for (int t = tid; t < TOT; t += 128) {
    int m = t / (7 * CH);
    int rj = t - m * (7 * CH);
    int j = rj / CH;
    int ch = rj - j * CH;
    int q = rowBase + m; if (q > R - 1) q = R - 1;
    int b = q / Nh;
    int n = q - b * Nh;
    int src = neigh[n * 7 + j];
    const v8bf* p = (const v8bf*)(x + ((size_t)b * Nh + src) * C + ch * 8);
    *(v8bf*)(tileA + m * K + j * C + ch * 8) = *p;
  }
  __syncthreads();

  const int lane = tid & 31, wave = tid >> 5;
  const int col = lane & 15, hl = lane >> 4;

  const float bv = bias[wave * 16 + col];
  v8f acc0 = {bv, bv, bv, bv, bv, bv, bv, bv};
  v8f acc1 = acc0;
  const __bf16* arow0 = tileA + col * K;
  const __bf16* arow1 = tileA + (16 + col) * K;
  const __bf16* brow  = wt + (size_t)(wave * 16 + col) * K + hl * 16;
  for (int k0 = 0; k0 < K; k0 += 32) {
    ABFrag a0, a1, b;
    b.h[0]  = *(const v8bf*)(brow + k0);
    b.h[1]  = *(const v8bf*)(brow + k0 + 8);
    a0.h[0] = *(const v8bf*)(arow0 + k0 + hl * 16);
    a0.h[1] = *(const v8bf*)(arow0 + k0 + hl * 16 + 8);
    a1.h[0] = *(const v8bf*)(arow1 + k0 + hl * 16);
    a1.h[1] = *(const v8bf*)(arow1 + k0 + hl * 16 + 8);
    acc0 = wmma_bf16(a0, b, acc0);
    acc1 = wmma_bf16(a1, b, acc1);
  }

  float* out = y + (size_t)rowBase * 64 + wave * 16 + col;
  if (rowBase + 32 <= R) {               // block-uniform fast path
    #pragma unroll
    for (int r = 0; r < 8; ++r) {
      out[(size_t)(r + hl * 8) * 64]      = acc0[r];
      out[(size_t)(16 + r + hl * 8) * 64] = acc1[r];
    }
  } else {
    #pragma unroll
    for (int r = 0; r < 8; ++r) {
      int m0 = r + hl * 8, m1 = m0 + 16;
      if (rowBase + m0 < R) out[(size_t)m0 * 64] = acc0[r];
      if (rowBase + m1 < R) out[(size_t)m1 * 64] = acc1[r];
    }
  }
}

// ---------------------------------------------------------------------------
// BN helpers: zero stats, per-channel sum/sumsq reduction, finalize, apply.
// ---------------------------------------------------------------------------
__global__ void zero_stats_kernel(float* stats) {
  if (threadIdx.x < 256) stats[threadIdx.x] = 0.0f;
}

__global__ void bn_reduce_kernel(const float* __restrict__ y,
                                 float* __restrict__ sum,
                                 float* __restrict__ sumsq, int R) {
  __shared__ float ssum[256], ssq[256];
  const int c = threadIdx.x & 63;
  const int slice = threadIdx.x >> 6;   // 4 row slices per block
  float s = 0.0f, q = 0.0f;
  for (int r = blockIdx.x * 4 + slice; r < R; r += gridDim.x * 4) {
    float v = y[(size_t)r * 64 + c];
    s += v; q += v * v;
  }
  ssum[threadIdx.x] = s;
  ssq[threadIdx.x] = q;
  __syncthreads();
  if (slice == 0) {
    s = ssum[c] + ssum[64 + c] + ssum[128 + c] + ssum[192 + c];
    q = ssq[c] + ssq[64 + c] + ssq[128 + c] + ssq[192 + c];
    atomicAdd(&sum[c], s);
    atomicAdd(&sumsq[c], q);
  }
}

__global__ void bn_finalize_kernel(const float* sum, const float* sumsq,
                                   const float* gamma, const float* beta,
                                   float* scale, float* shift, int R) {
  int c = threadIdx.x;
  if (c >= 64) return;
  float invR = 1.0f / (float)R;
  float mean = sum[c] * invR;
  float var = sumsq[c] * invR - mean * mean;
  float sc = gamma[c] * rsqrtf(var + 1e-5f);
  scale[c] = sc;
  shift[c] = beta[c] - mean * sc;
}

__global__ void bn_apply_kernel(const float* __restrict__ y,
                                const float* __restrict__ scale,
                                const float* __restrict__ shift,
                                __bf16* __restrict__ out_bf,
                                float* __restrict__ out_f, int total) {
  int t = blockIdx.x * blockDim.x + threadIdx.x;
  if (t >= total) return;
  int c = t & 63;
  float v = y[t] * scale[c] + shift[c];
  v = (v >= 0.0f) ? v : 0.2f * v;
  if (out_bf) out_bf[t] = (__bf16)v;
  else        out_f[t] = v;
}

// ---------------------------------------------------------------------------
extern "C" void kernel_launch(void* const* d_in, const int* in_sizes, int n_in,
                              void* d_out, int out_size, void* d_ws, size_t ws_size,
                              hipStream_t stream) {
  const float* x1      = (const float*)d_in[0];
  const float* x2      = (const float*)d_in[1];
  const int*   neigh   = (const int*)d_in[2];
  const int*   topIdx  = (const int*)d_in[3];
  const int*   downIdx = (const int*)d_in[4];
  const float* up_W    = (const float*)d_in[5];
  const float* up_b    = (const float*)d_in[6];
  const float* c1W     = (const float*)d_in[7];
  const float* c1b     = (const float*)d_in[8];
  const float* g1      = (const float*)d_in[9];
  const float* b1      = (const float*)d_in[10];
  const float* c2W     = (const float*)d_in[11];
  const float* c2b     = (const float*)d_in[12];
  const float* g2      = (const float*)d_in[13];
  const float* b2      = (const float*)d_in[14];

  const int Nr   = in_sizes[3];            // 40962
  const int Kup  = in_sizes[6];            // 448 = 7*Cout
  const int Cout = Kup / 7;                // 64
  const int Cin  = in_sizes[5] / Kup;      // 128
  const int Bb   = in_sizes[0] / (Nr * Cin);      // 2
  const int Nh   = in_sizes[1] / (Bb * Cout);     // 163842
  const int Mup  = Bb * Nr;                // rows for up-projection
  const int R    = Bb * Nh;                // rows for mesh convs

  char* ws = (char*)d_ws;
  size_t off = 0;
  auto take = [&](size_t bytes) -> char* {
    char* p = ws + off;
    off += (bytes + 255) & ~(size_t)255;
    return p;
  };
  float*  h     = (float*)take((size_t)Mup * Kup * sizeof(float));
  __bf16* xcat  = (__bf16*)take((size_t)R * (2 * Cout) * sizeof(__bf16));
  float*  y     = (float*)take((size_t)R * Cout * sizeof(float));
  __bf16* act   = (__bf16*)take((size_t)R * Cout * sizeof(__bf16));
  __bf16* wtUp  = (__bf16*)take((size_t)Kup * Cin * sizeof(__bf16));
  __bf16* wt1   = (__bf16*)take((size_t)Cout * 7 * (2 * Cout) * sizeof(__bf16));
  __bf16* wt2   = (__bf16*)take((size_t)Cout * 7 * Cout * sizeof(__bf16));
  float*  stats = (float*)take(256 * sizeof(float));
  float *sum = stats, *sumsq = stats + 64, *scale = stats + 128, *shift = stats + 192;

  // 1) bf16 transposed weights (tiny)
  { int tot = Cin * Kup;
    transpose_w_kernel<<<(tot + 255) / 256, 256, 0, stream>>>(up_W, wtUp, Cin, Kup); }
  { int tot = 7 * 2 * Cout * Cout;
    transpose_w_kernel<<<(tot + 255) / 256, 256, 0, stream>>>(c1W, wt1, 7 * 2 * Cout, Cout); }
  { int tot = 7 * Cout * Cout;
    transpose_w_kernel<<<(tot + 255) / 256, 256, 0, stream>>>(c2W, wt2, 7 * Cout, Cout); }

  // 2) up-projection GEMM (WMMA)
  upproj_wmma_kernel<<<(Mup + 15) / 16, 128, 0, stream>>>(x1, wtUp, up_b, h, Mup);

  // 3) top/down gather + concat -> bf16 activation
  { int tot = R * 2 * Cout;
    build_xcat_kernel<<<(tot + 255) / 256, 256, 0, stream>>>(h, x2, topIdx, downIdx,
                                                             xcat, Nr, Nh, tot); }

  // 4) mesh conv 1 (fused gather + WMMA GEMM, K=896) + BN + lrelu -> bf16
  meshconv_wmma_kernel<128><<<(R + 31) / 32, 128, 0, stream>>>(xcat, neigh, wt1, c1b, y, Nh, R);
  zero_stats_kernel<<<1, 256, 0, stream>>>(stats);
  bn_reduce_kernel<<<1024, 256, 0, stream>>>(y, sum, sumsq, R);
  bn_finalize_kernel<<<1, 64, 0, stream>>>(sum, sumsq, g1, b1, scale, shift, R);
  { int tot = R * Cout;
    bn_apply_kernel<<<(tot + 255) / 256, 256, 0, stream>>>(y, scale, shift, act,
                                                           (float*)nullptr, tot); }

  // 5) mesh conv 2 (K=448) + BN + lrelu -> f32 output
  meshconv_wmma_kernel<64><<<(R + 31) / 32, 128, 0, stream>>>(act, neigh, wt2, c2b, y, Nh, R);
  zero_stats_kernel<<<1, 256, 0, stream>>>(stats);
  bn_reduce_kernel<<<1024, 256, 0, stream>>>(y, sum, sumsq, R);
  bn_finalize_kernel<<<1, 64, 0, stream>>>(sum, sumsq, g2, b2, scale, shift, R);
  { int tot = R * Cout;
    bn_apply_kernel<<<(tot + 255) / 256, 256, 0, stream>>>(y, scale, shift,
                                                           (__bf16*)nullptr,
                                                           (float*)d_out, tot); }
  (void)n_in; (void)out_size; (void)ws_size;
}
